// VQVAE_57037165691152
// MI455X (gfx1250) — compile-verified, matrix operand order checked by reading, not run on hardware
//
#include <hip/hip_runtime.h>
#include <hip/hip_bf16.h>

typedef __bf16 bf16_t;
typedef bf16_t v16bf __attribute__((ext_vector_type(16)));
typedef float  v8f   __attribute__((ext_vector_type(8)));
typedef unsigned short u16;
typedef u16 u16x8 __attribute__((ext_vector_type(8)));

#define ACT_NONE  0
#define ACT_LRELU 1
#define ACT_RELU  2
#define ACT_TANH  3

__device__ __forceinline__ u16 f2bf(float f) {
    unsigned int u = __builtin_bit_cast(unsigned int, f);
    u += 0x7fffu + ((u >> 16) & 1u);           // round-to-nearest-even
    return (u16)(u >> 16);
}
__device__ __forceinline__ float bf2f(u16 s) {
    unsigned int u = ((unsigned int)s) << 16;
    return __builtin_bit_cast(float, u);
}
__device__ __forceinline__ v16bf mk16(u16x8 lo, u16x8 hi) {
    union { u16x8 p[2]; v16bf v; } u;
    u.p[0] = lo; u.p[1] = hi;
    return u.v;
}

// ---------------------------------------------------------------------------
// elementwise helpers
// ---------------------------------------------------------------------------
__global__ __launch_bounds__(256)
void x_nchw_to_nhwc_bf16_kernel(const float* __restrict__ src, u16* __restrict__ dst,
                                int C, int HW, size_t total) {
    size_t i = (size_t)blockIdx.x * blockDim.x + threadIdx.x;   // NCHW linear
    if (i >= total) return;
    int p = (int)(i % HW);
    size_t t = i / HW;
    int c = (int)(t % C);
    size_t b = t / C;
    dst[(b * HW + p) * C + c] = f2bf(src[i]);
}

__global__ __launch_bounds__(256)
void lrelu_bf16_kernel(u16* __restrict__ p, size_t n) {
    size_t i = (size_t)blockIdx.x * blockDim.x + threadIdx.x;
    if (i < n) {
        float v = bf2f(p[i]);
        p[i] = f2bf(v > 0.f ? v : 0.01f * v);
    }
}

// ---------------------------------------------------------------------------
// Weight repack: fp32 OIHW (or ConvT IOHW + spatial flip) -> bf16 [Cout][KH*KW][Cin]
// ---------------------------------------------------------------------------
__global__ __launch_bounds__(256)
void repack_w_kernel(const float* __restrict__ src, u16* __restrict__ dst,
                     int Cout, int Cin, int KH, int KW, int flipT) {
    int KHW = KH * KW;
    size_t total = (size_t)Cout * Cin * KHW;
    size_t i = (size_t)blockIdx.x * blockDim.x + threadIdx.x;
    if (i >= total) return;
    int ci  = (int)(i % Cin);
    size_t t = i / Cin;
    int khw = (int)(t % KHW);
    int co  = (int)(t / KHW);
    float v;
    if (flipT) {  // ConvTranspose weight [Cin][Cout][KH][KW]; flip spatially
        int kh = khw / KW, kw = khw - kh * KW;
        v = src[((size_t)ci * Cout + co) * KHW + (size_t)(KH - 1 - kh) * KW + (KW - 1 - kw)];
    } else {      // [Cout][Cin][KH][KW]
        v = src[((size_t)co * Cin + ci) * KHW + khw];
    }
    dst[i] = f2bf(v);
}

// ---------------------------------------------------------------------------
// Implicit-GEMM convolution on the WMMA pipe, NHWC activations.
//   GEMM: M = Cout, N = B*Ho*Wo pixels, K = KH*KW*CIN.
//   NHWC => B-fragment K-slots are channel-contiguous: 2 aligned b128 loads
//   per fragment, same as A. Each wave: 2 M-tiles x 2 N-tiles, 4 WMMAs/step.
//   Out-of-range taps redirect the base pointer to a zero page.
//   LDIL=1: lhs-dilation-by-2 gather (ConvTranspose k4 s2 p1 as normal conv).
//   ACT_TANH epilogue writes NCHW fp32 (final recon) instead of NHWC bf16.
// ---------------------------------------------------------------------------
template<int CIN, int TH, int TW, int LDIL>
__global__ __launch_bounds__(256)
void wmma_conv_kernel(const u16* __restrict__ act,   // NHWC bf16
                      const u16* __restrict__ wgt,   // [Cout][KHW][CIN] bf16
                      const float* __restrict__ bias,
                      const u16* __restrict__ resid, // NHWC bf16 (out-shaped) or null
                      const u16* __restrict__ zpage,
                      u16* __restrict__ out_bf,      // NHWC bf16
                      float* __restrict__ out_f,     // NCHW fp32 (ACT_TANH only)
                      int B, int Cout, int Ho, int Wo,
                      int KH, int KW, int stride, int pad, int actm) {
    const int lane  = threadIdx.x & 31;
    const int wid   = threadIdx.x >> 5;
    const int lcol  = lane & 15;
    const int hiOff = (lane < 16) ? 0 : 8;
    const int KHW   = KH * KW;
    const int Kdim  = KHW * CIN;
    const int Npix  = B * Ho * Wo;

    // two N columns (output pixels) per lane
    int n0 = (blockIdx.x * 8 + wid) * 32 + lcol;
    int n1 = n0 + 16;
    bool ok0 = (n0 < Npix), ok1 = (n1 < Npix);
    int nc0 = ok0 ? n0 : (Npix - 1), nc1 = ok1 ? n1 : (Npix - 1);
    int b0 = nc0 / (Ho * Wo); int r0 = nc0 - b0 * (Ho * Wo);
    int oh0 = r0 / Wo, ow0 = r0 - oh0 * Wo;
    int b1 = nc1 / (Ho * Wo); int r1 = nc1 - b1 * (Ho * Wo);
    int oh1 = r1 / Wo, ow1 = r1 - oh1 * Wo;
    int ih00 = oh0 * stride - pad, iw00 = ow0 * stride - pad;
    int ih10 = oh1 * stride - pad, iw10 = ow1 * stride - pad;

    // two M tiles per wave
    int m0 = blockIdx.y * 32 + lcol;
    int m1 = m0 + 16;
    const u16* wrow0 = wgt + (size_t)((m0 < Cout) ? m0 : 0) * Kdim;
    const u16* wrow1 = wgt + (size_t)((m1 < Cout) ? m1 : 0) * Kdim;
    const u16* zp    = zpage + hiOff;

    v8f acc00 = {}, acc01 = {}, acc10 = {}, acc11 = {};

    for (int khw = 0; khw < KHW; ++khw) {
        int kh = khw / KW;
        int kw = khw - kh * KW;

        auto bptr = [&](bool ok, int bb, int ihl, int iwl) -> const u16* {
            bool sv; int ih, iw;
            if (LDIL) {
                sv = ((ihl | iwl) >= 0) && (ihl < 2 * TH - 1) && (iwl < 2 * TW - 1)
                     && (((ihl | iwl) & 1) == 0);
                ih = ihl >> 1; iw = iwl >> 1;
            } else {
                sv = ((ihl | iwl) >= 0) && (ihl < TH) && (iwl < TW);
                ih = ihl; iw = iwl;
            }
            sv = sv && ok;
            return sv ? (act + ((size_t)(bb * TH + ih) * TW + iw) * CIN + hiOff) : zp;
        };
        const u16* bp0 = bptr(ok0, b0, ih00 + kh, iw00 + kw);
        const u16* bp1 = bptr(ok1, b1, ih10 + kh, iw10 + kw);
        const u16* a0  = wrow0 + khw * CIN + hiOff;
        const u16* a1  = wrow1 + khw * CIN + hiOff;

        for (int c0 = 0; c0 < CIN; c0 += 32) {
            v16bf af0, af1, bf0, bf1;
            if constexpr (CIN >= 64) {
                __builtin_prefetch(bp0 + 32, 0, 1);
                __builtin_prefetch(bp1 + 32, 0, 1);
                // all bases 16B aligned by construction: explicit b128 loads
                const u16x8* A0 = (const u16x8*)a0;
                const u16x8* A1 = (const u16x8*)a1;
                const u16x8* B0 = (const u16x8*)bp0;
                const u16x8* B1 = (const u16x8*)bp1;
                af0 = mk16(A0[0], A0[2]);      // K slots 0..7 and 16..23 (+hiOff)
                af1 = mk16(A1[0], A1[2]);
                bf0 = mk16(B0[0], B0[2]);
                bf1 = mk16(B1[0], B1[2]);
            } else {  // Cin=3 stem: masked scalar path
#pragma unroll
                for (int i = 0; i < 16; ++i) {
                    const int off = (i < 8) ? i : (i + 8);
                    u16 A0v = a0[off], A1v = a1[off];
                    u16 B0v = bp0[off], B1v = bp1[off];
                    bool cv = (c0 + off + hiOff) < CIN;
                    B0v = cv ? B0v : (u16)0;
                    B1v = cv ? B1v : (u16)0;
                    af0[i] = __builtin_bit_cast(bf16_t, A0v);
                    af1[i] = __builtin_bit_cast(bf16_t, A1v);
                    bf0[i] = __builtin_bit_cast(bf16_t, B0v);
                    bf1[i] = __builtin_bit_cast(bf16_t, B1v);
                }
            }
            a0 += 32; a1 += 32; bp0 += 32; bp1 += 32;
            acc00 = __builtin_amdgcn_wmma_f32_16x16x32_bf16(false, af0, false, bf0, (short)0, acc00, false, false);
            acc10 = __builtin_amdgcn_wmma_f32_16x16x32_bf16(false, af1, false, bf0, (short)0, acc10, false, false);
            acc01 = __builtin_amdgcn_wmma_f32_16x16x32_bf16(false, af0, false, bf1, (short)0, acc01, false, false);
            acc11 = __builtin_amdgcn_wmma_f32_16x16x32_bf16(false, af1, false, bf1, (short)0, acc11, false, false);
        }
    }

    // C/D layout: VGPR r, lane l -> M = r + 8*(l>>4), N = l&15
    int mtbase = blockIdx.y * 32 + ((lane >> 4) << 3);   // multiple of 8

    if (actm == ACT_TANH) {   // final layer: NCHW fp32 recon
#pragma unroll
        for (int t = 0; t < 2; ++t)
#pragma unroll
        for (int r = 0; r < 8; ++r) {
            int mm = mtbase + t * 16 + r;
            if (mm >= Cout) continue;
#pragma unroll
            for (int u = 0; u < 2; ++u) {
                if (!(u ? ok1 : ok0)) continue;
                float v = u ? (t ? acc11[r] : acc01[r]) : (t ? acc10[r] : acc00[r]);
                if (bias) v += bias[mm];
                int bb = u ? b1 : b0, oh = u ? oh1 : oh0, ow = u ? ow1 : ow0;
                out_f[(((size_t)bb * Cout + mm) * Ho + oh) * Wo + ow] = tanhf(v);
            }
        }
        return;
    }

    // NHWC bf16 epilogue: 8 consecutive couts -> one b128 store
#pragma unroll
    for (int t = 0; t < 2; ++t) {
        int mbase = mtbase + t * 16;
        if (mbase >= Cout) continue;
#pragma unroll
        for (int u = 0; u < 2; ++u) {
            if (!(u ? ok1 : ok0)) continue;
            int pix = u ? nc1 : nc0;
            size_t obase = (size_t)pix * Cout + mbase;
            u16x8 rv = {};
            if (resid) rv = *(const u16x8*)(resid + obase);
            u16x8 ov;
#pragma unroll
            for (int r = 0; r < 8; ++r) {
                float v = u ? (t ? acc11[r] : acc01[r]) : (t ? acc10[r] : acc00[r]);
                if (bias) v += bias[mbase + r];
                if (resid) v += bf2f(rv[r]);
                if (actm == ACT_LRELU)      v = v > 0.f ? v : 0.01f * v;
                else if (actm == ACT_RELU)  v = v > 0.f ? v : 0.f;
                ov[r] = f2bf(v);
            }
            *(u16x8*)(out_bf + obase) = ov;
        }
    }
}

template<int CIN, int TH, int TW, int LDIL>
static inline void conv_launch(const u16* act, const u16* wgt, const float* bias,
                               const u16* resid, const u16* zp,
                               u16* out_bf, float* out_f,
                               int B, int Cout, int KH, int KW, int s, int p, int actm,
                               hipStream_t stream) {
    int Ho, Wo;
    if (LDIL) { Ho = 2 * TH; Wo = 2 * TW; }
    else { Ho = (TH + 2 * p - KH) / s + 1; Wo = (TW + 2 * p - KW) / s + 1; }
    int Npix = B * Ho * Wo;
    dim3 g((unsigned)((Npix + 255) / 256), (unsigned)((Cout + 31) / 32));
    wmma_conv_kernel<CIN, TH, TW, LDIL><<<g, 256, 0, stream>>>(
        act, wgt, bias, resid, zp, out_bf, out_f, B, Cout, Ho, Wo, KH, KW, s, p, actm);
}

// ---------------------------------------------------------------------------
// Vector quantization (fp32 scalar; NHWC makes vectors contiguous)
// ---------------------------------------------------------------------------
__global__ __launch_bounds__(256)
void ek2_kernel(const float* __restrict__ E, float* __restrict__ ek2, int Kn, int D) {
    int k = blockIdx.x * blockDim.x + threadIdx.x;
    if (k >= Kn) return;
    float s = 0.f;
    for (int d = 0; d < D; ++d) { float e = E[(size_t)k * D + d]; s += e * e; }
    ek2[k] = s;
}

__global__ __launch_bounds__(256)
void vq_kernel(const u16* __restrict__ z,       // [N, 64] bf16 (NHWC latents)
               const float* __restrict__ E,     // [Kn, 64]
               const float* __restrict__ ek2,   // [Kn]
               u16* __restrict__ zq,            // [N, 64] bf16
               float* __restrict__ lossAcc,
               int N, int Kn) {
    constexpr int D = 64;
    int nidx = blockIdx.x * blockDim.x + threadIdx.x;
    if (nidx >= N) return;
    size_t base = (size_t)nidx * D;

    float zn[D];
#pragma unroll
    for (int d = 0; d < D; ++d) zn[d] = bf2f(z[base + d]);

    float best = 3.4e38f; int bi = 0;
    for (int k = 0; k < Kn; ++k) {
        const float* e = E + (size_t)k * D;
        float dot = 0.f;
#pragma unroll 8
        for (int d = 0; d < D; ++d) dot += zn[d] * e[d];
        float dist = ek2[k] - 2.f * dot;
        if (dist < best) { best = dist; bi = k; }
    }
    const float* e = E + (size_t)bi * D;
    float sse = 0.f;
#pragma unroll
    for (int d = 0; d < D; ++d) {
        float q = e[d];
        float df = q - zn[d];
        sse += df * df;
        zq[base + d] = f2bf(q);
    }
    atomicAdd(lossAcc, sse);
}

__global__ void finalize_loss_kernel(const float* __restrict__ lossAcc,
                                     float* __restrict__ dst, float scale) {
    if (threadIdx.x == 0 && blockIdx.x == 0) dst[0] = lossAcc[0] * scale;
}

// ---------------------------------------------------------------------------
// Host orchestration
// ---------------------------------------------------------------------------
extern "C" void kernel_launch(void* const* d_in, const int* in_sizes, int n_in,
                              void* d_out, int out_size, void* d_ws, size_t ws_size,
                              hipStream_t stream) {
    (void)in_sizes; (void)n_in; (void)ws_size;
    const int B = 64;

    char* ws = (char*)d_ws;
    size_t off = 0;
    auto allocB = [&](size_t bytes) -> char* {
        char* p = ws + off;
        off += (bytes + 255) & ~(size_t)255;
        return p;
    };
    float* lossAcc = (float*)allocB(4);
    float* ek2     = (float*)allocB(512 * sizeof(float));
    auto allocU = [&](size_t n) { return (u16*)allocB(n * sizeof(u16)); };

    // zero page for out-of-range im2col taps (NHWC: only ~Cin elements needed)
    u16* zp = allocU(4096);
    hipMemsetAsync(zp, 0, 8192, stream);

    auto repack = [&](int i, int Cout, int Cin, int KH, int KW, int flipT) {
        size_t n = (size_t)Cout * Cin * KH * KW;
        u16* p = allocU(n);
        repack_w_kernel<<<dim3((unsigned)((n + 255) / 256)), 256, 0, stream>>>(
            (const float*)d_in[i], p, Cout, Cin, KH, KW, flipT);
        return p;
    };

    // --- weights ---
    u16* w_down0 = repack(1, 64, 3, 4, 4, 0);
    u16* w_down1 = repack(3, 128, 64, 4, 4, 0);
    u16* w_down2 = repack(5, 256, 128, 4, 4, 0);
    u16* w_mid   = repack(7, 256, 256, 3, 3, 0);
    u16 *w_er1[6], *w_er2[6], *w_dr1[6], *w_dr2[6];
    for (int i = 0; i < 6; ++i) {
        w_er1[i] = repack(9 + 2 * i, 256, 256, 3, 3, 0);
        w_er2[i] = repack(10 + 2 * i, 256, 256, 1, 1, 0);
    }
    u16* w_head = repack(21, 64, 256, 1, 1, 0);
    u16* w_din  = repack(24, 256, 64, 3, 3, 0);
    for (int i = 0; i < 6; ++i) {
        w_dr1[i] = repack(26 + 2 * i, 256, 256, 3, 3, 0);
        w_dr2[i] = repack(27 + 2 * i, 256, 256, 1, 1, 0);
    }
    u16* w_up0 = repack(38, 128, 256, 4, 4, 1);
    u16* w_up1 = repack(40, 64, 128, 4, 4, 1);
    u16* w_up2 = repack(42, 3, 64, 4, 4, 1);

    // --- activation buffers (bf16, NHWC) ---
    const size_t BIG = (size_t)B * 64 * 64 * 64 + 64;   // +tail margin
    u16* buf0 = allocU(BIG);
    u16* buf1 = allocU(BIG);
    u16* buf2 = allocU(BIG);
    u16* zbuf  = allocU((size_t)B * 16 * 16 * 64);
    u16* zqbuf = allocU((size_t)B * 16 * 16 * 64);

    auto lrelu_ip = [&](u16* p, size_t n) {
        lrelu_bf16_kernel<<<dim3((unsigned)((n + 255) / 256)), 256, 0, stream>>>(p, n);
    };

    // ===================== encoder =====================
    x_nchw_to_nhwc_bf16_kernel<<<dim3((unsigned)((3145728 + 255) / 256)), 256, 0, stream>>>(
        (const float*)d_in[0], buf0, 3, 128 * 128, (size_t)3145728);
    conv_launch<3, 128, 128, 0>(buf0, w_down0, (const float*)d_in[2], nullptr, zp, buf1, nullptr,
                                B, 64, 4, 4, 2, 1, ACT_LRELU, stream);
    conv_launch<64, 64, 64, 0>(buf1, w_down1, (const float*)d_in[4], nullptr, zp, buf2, nullptr,
                               B, 128, 4, 4, 2, 1, ACT_LRELU, stream);
    conv_launch<128, 32, 32, 0>(buf2, w_down2, (const float*)d_in[6], nullptr, zp, buf0, nullptr,
                                B, 256, 4, 4, 2, 1, ACT_LRELU, stream);
    conv_launch<256, 16, 16, 0>(buf0, w_mid, (const float*)d_in[8], nullptr, zp, buf1, nullptr,
                                B, 256, 3, 3, 1, 1, ACT_LRELU, stream);

    u16 *cur = buf1, *spare = buf0, *tmp = buf2;
    for (int i = 0; i < 6; ++i) {
        conv_launch<256, 16, 16, 0>(cur, w_er1[i], nullptr, nullptr, zp, tmp, nullptr,
                                    B, 256, 3, 3, 1, 1, ACT_RELU, stream);
        conv_launch<256, 16, 16, 0>(tmp, w_er2[i], nullptr, cur, zp, spare, nullptr,
                                    B, 256, 1, 1, 1, 0, ACT_NONE, stream);
        u16* t = cur; cur = spare; spare = t;
    }
    lrelu_ip(cur, (size_t)B * 256 * 16 * 16);
    conv_launch<256, 16, 16, 0>(cur, w_head, (const float*)d_in[22], nullptr, zp, zbuf, nullptr,
                                B, 64, 1, 1, 1, 0, ACT_LRELU, stream);

    // ===================== vector quantization =====================
    hipMemsetAsync(lossAcc, 0, sizeof(float), stream);
    ek2_kernel<<<dim3(2), 256, 0, stream>>>((const float*)d_in[23], ek2, 512, 64);
    vq_kernel<<<dim3(64), 256, 0, stream>>>(zbuf, (const float*)d_in[23], ek2, zqbuf,
                                            lossAcc, 16384, 512);

    // ===================== decoder =====================
    conv_launch<64, 16, 16, 0>(zqbuf, w_din, (const float*)d_in[25], nullptr, zp, buf0, nullptr,
                               B, 256, 3, 3, 1, 1, ACT_LRELU, stream);
    cur = buf0; spare = buf1; tmp = buf2;
    for (int i = 0; i < 6; ++i) {
        conv_launch<256, 16, 16, 0>(cur, w_dr1[i], nullptr, nullptr, zp, tmp, nullptr,
                                    B, 256, 3, 3, 1, 1, ACT_RELU, stream);
        conv_launch<256, 16, 16, 0>(tmp, w_dr2[i], nullptr, cur, zp, spare, nullptr,
                                    B, 256, 1, 1, 1, 0, ACT_NONE, stream);
        u16* t = cur; cur = spare; spare = t;
    }
    lrelu_ip(cur, (size_t)B * 256 * 16 * 16);

    // up convs (ConvTranspose2d k4 s2 p1) on the WMMA pipe via lhs-dilation
    conv_launch<256, 16, 16, 1>(cur, w_up0, (const float*)d_in[39], nullptr, zp, buf2, nullptr,
                                B, 128, 4, 4, 1, 2, ACT_LRELU, stream);
    u16* u1out = (cur == buf0) ? buf1 : buf0;
    conv_launch<128, 32, 32, 1>(buf2, w_up1, (const float*)d_in[41], nullptr, zp, u1out, nullptr,
                                B, 64, 4, 4, 1, 2, ACT_LRELU, stream);
    conv_launch<64, 64, 64, 1>(u1out, w_up2, (const float*)d_in[43], nullptr, zp, nullptr,
                               (float*)d_out, B, 3, 4, 4, 1, 2, ACT_TANH, stream);

    // loss scalar at the tail of d_out: 1.25 * SSE / (N*D)
    finalize_loss_kernel<<<dim3(1), 32, 0, stream>>>(
        lossAcc, (float*)d_out + (out_size - 1), 1.25f / (16384.f * 64.f));
}
